// SequenceLabelingLSTM_CRF_17308718203177
// MI455X (gfx1250) — compile-verified, matrix operand order checked by reading, not run on hardware
//
#include <hip/hip_runtime.h>
#include <hip/hip_bf16.h>
#include <hip/hip_fp16.h>

#define BDIM 64
#define TDIM 4096
#define HDIM 128
#define CDIM 7

typedef _Float16 v16h __attribute__((ext_vector_type(16)));
typedef _Float16 v8h  __attribute__((ext_vector_type(8)));
typedef float    v8f  __attribute__((ext_vector_type(8)));

union HFrag {
    v16h v;
    v8h  h8[2];
    _Float16 e[16];
};

// ---- fast transcendentals on the recurrence critical path -----------------
// v_exp_f32 computes exp2; v_rcp_f32 is ~1ulp.  Both saturate correctly:
//   exp2(+big)=inf -> rcp -> 0 ;  exp2(-big)=0  -> exact limits at +/-1.
#define LOG2E  1.4426950408889634f

__device__ __forceinline__ float sigf(float x) {
    const float e = __builtin_amdgcn_exp2f(x * -LOG2E);          // exp(-x)
    return __builtin_amdgcn_rcpf(1.0f + e);
}

__device__ __forceinline__ float tanhf_fast(float x) {
#if __has_builtin(__builtin_amdgcn_tanhf)
    return __builtin_amdgcn_tanhf(x);                            // v_tanh_f32
#elif __has_builtin(__builtin_amdgcn_tanh_f32)
    return __builtin_amdgcn_tanh_f32(x);
#else
    const float e = __builtin_amdgcn_exp2f(x * (2.0f * LOG2E));  // exp(2x)
    return 1.0f - 2.0f * __builtin_amdgcn_rcpf(e + 1.0f);
#endif
}

// ---------------------------------------------------------------------------
// Kernel 1: fused frontend.  One thread per (b,t):
//   causal conv (pad 2, k=3) on X_L/X_R -> relu -> MLP -> softmax gate ->
//   tilde -> residual -> layernorm(16) -> f16 tilde_X to workspace.
// ---------------------------------------------------------------------------
__global__ __launch_bounds__(256)
void frontend_kernel(const float* __restrict__ x,
                     const float* __restrict__ conv_w,   // (16,8,3)
                     const float* __restrict__ conv_b,   // (16)
                     const float* __restrict__ mlp_w1,   // (16,32)
                     const float* __restrict__ mlp_b1,   // (16)
                     const float* __restrict__ mlp_w2,   // (2,16)
                     const float* __restrict__ mlp_b2,   // (2)
                     const float* __restrict__ ln_g,     // (16)
                     const float* __restrict__ ln_b,     // (16)
                     _Float16* __restrict__ tx)          // (B,T,16) f16
{
    const int idx = blockIdx.x * blockDim.x + threadIdx.x;   // b*T + t
    const int b = idx / TDIM;
    const int t = idx - b * TDIM;

    // rows x[b, t-2 .. t, :], zero-padded on the left (causal pad = 2)
    float xr[3][16];
#pragma unroll
    for (int k = 0; k < 3; ++k) {
        const int tt = t - 2 + k;
        if (tt >= 0) {
            const float* row = x + ((size_t)b * TDIM + tt) * 16;
#pragma unroll
            for (int c = 0; c < 16; ++c) xr[k][c] = row[c];
        } else {
#pragma unroll
            for (int c = 0; c < 16; ++c) xr[k][c] = 0.0f;
        }
    }

    // causal conv + relu on left/right halves -> H_cat(32)
    float hcat[32];
#pragma unroll
    for (int p = 0; p < 16; ++p) {
        float aL = conv_b[p], aR = conv_b[p];
#pragma unroll
        for (int ci = 0; ci < 8; ++ci) {
#pragma unroll
            for (int k = 0; k < 3; ++k) {
                const float w = conv_w[(p * 8 + ci) * 3 + k];
                aL += w * xr[k][ci];
                aR += w * xr[k][8 + ci];
            }
        }
        hcat[p]      = fmaxf(aL, 0.0f);
        hcat[16 + p] = fmaxf(aR, 0.0f);
    }

    // MLP hidden(16) -> logits(2) -> softmax
    float hid[16];
#pragma unroll
    for (int p = 0; p < 16; ++p) {
        float s = mlp_b1[p];
#pragma unroll
        for (int q = 0; q < 32; ++q) s += mlp_w1[p * 32 + q] * hcat[q];
        hid[p] = fmaxf(s, 0.0f);
    }
    float l0 = mlp_b2[0], l1 = mlp_b2[1];
#pragma unroll
    for (int p = 0; p < 16; ++p) {
        l0 += mlp_w2[p] * hid[p];
        l1 += mlp_w2[16 + p] * hid[p];
    }
    const float mx = fmaxf(l0, l1);
    const float e0 = __builtin_amdgcn_exp2f((l0 - mx) * LOG2E);
    const float e1 = __builtin_amdgcn_exp2f((l1 - mx) * LOG2E);
    const float rs = __builtin_amdgcn_rcpf(e0 + e1);
    const float a0 = e0 * rs, a1 = e1 * rs;

    // tilde + residual -> layernorm over 16
    float z[16];
#pragma unroll
    for (int c = 0; c < 8; ++c) {
        z[c]     = xr[2][c]     * (1.0f + a0);
        z[8 + c] = xr[2][8 + c] * (1.0f + a1);
    }
    float mu = 0.0f;
#pragma unroll
    for (int c = 0; c < 16; ++c) mu += z[c];
    mu *= (1.0f / 16.0f);
    float var = 0.0f;
#pragma unroll
    for (int c = 0; c < 16; ++c) { const float d = z[c] - mu; var += d * d; }
    var *= (1.0f / 16.0f);
    const float rstd = __frsqrt_rn(var + 1e-5f);

    _Float16* orow = tx + (size_t)idx * 16;
#pragma unroll
    for (int c = 0; c < 16; ++c)
        orow[c] = (_Float16)((z[c] - mu) * rstd * ln_g[c] + ln_b[c]);
}

// ---------------------------------------------------------------------------
// Kernel 2/3: persistent LSTM layer.
//   grid = 4 workgroups (16 batch rows each), 512 threads = 16 waves.
//   Concatenated weights [W_hh^T | W_ih^T] packed once into per-wave WMMA
//   B-fragments (KCH chunks of K=32).  Each step:
//     stage x_t into LDS A-buffer -> v_wmma_f32_16x16x32_f16 (f32 acc,
//     bias-initialized) -> gates to LDS -> sigmoid/tanh recurrence with
//     c-state in thread registers -> new h back into A-buffer + f16 output.
//   Fragment index maps (ISA 7.12.2, 16-bit A 16x32 / B 32x16):
//     k = 16*(h>>3) + 8*(lane>>4) + (h&7),  row/col = lane&15.
// ---------------------------------------------------------------------------
template <int KCH, int XDIM>
__global__ __launch_bounds__(512)
void lstm_layer_kernel(const _Float16* __restrict__ xin,  // (B,T,XDIM) f16
                       const float* __restrict__ w_ih,    // (512, XDIM)
                       const float* __restrict__ w_hh,    // (512, 128)
                       const float* __restrict__ bias,    // (512)
                       _Float16* __restrict__ yout)       // (B,T,128) f16
{
    __shared__ _Float16 hbuf[16][256];    // A staging: cols [0,128)=h, [128,..)=x, pad=0
    __shared__ float    gbuf[16][512];    // gate accumulators (f32)

    const int tid  = threadIdx.x;
    const int lane = tid & 31;
    const int wv   = tid >> 5;            // wave 0..15, owns gate cols [wv*32, wv*32+32)
    const int bbase = blockIdx.x * 16;    // batch slice

    // zero the whole A buffer (h starts at 0, K-pad columns stay 0 forever)
    for (int i = tid; i < 16 * 256; i += 512)
        ((_Float16*)hbuf)[i] = (_Float16)0.0f;
    __syncthreads();

    const int frow = lane & 15;           // A row m  /  B column n_local
    const int kb8  = (lane >> 4) * 8;     // per-lane k base inside a chunk

    // ---- pack weights into register-resident B fragments (once) ----
    HFrag bfrag[2][KCH];
    float bs[2];
#pragma unroll
    for (int tile = 0; tile < 2; ++tile) {
        const int n = wv * 32 + tile * 16 + frow;   // gate column 0..511
        bs[tile] = bias[n];
#pragma unroll
        for (int kc = 0; kc < KCH; ++kc) {
#pragma unroll
            for (int h = 0; h < 16; ++h) {
                const int k = kc * 32 + ((h >> 3) << 4) + kb8 + (h & 7);
                float wval;
                if (k < 128)              wval = w_hh[n * 128 + k];
                else if (k < 128 + XDIM)  wval = w_ih[n * XDIM + (k - 128)];
                else                      wval = 0.0f;
                bfrag[tile][kc].e[h] = (_Float16)wval;
            }
        }
    }

    // cell state lives in registers: thread owns elems e = tid + 512*q
    float cst[4] = {0.0f, 0.0f, 0.0f, 0.0f};
    const int mtop = (lane >> 4) * 8;     // C/D fragment row base

#pragma unroll 1
    for (int t = 0; t < TDIM; ++t) {
        // stage x_t into A-buffer columns [128, 128+XDIM)
        for (int i = tid; i < 16 * XDIM; i += 512) {
            const int r = i / XDIM, c = i - r * XDIM;
            hbuf[r][128 + c] = xin[((size_t)(bbase + r) * TDIM + t) * XDIM + c];
        }
        if (t + 1 < TDIM && tid < 16)     // hint next step's rows into cache
            __builtin_prefetch(&xin[((size_t)(bbase + tid) * TDIM + (t + 1)) * XDIM], 0, 0);
        __syncthreads();

        // gates = [h_t | x_t | 0] @ [W_hh^T ; W_ih^T ; 0] + b   (f32 accum)
        v8f acc0, acc1;
#pragma unroll
        for (int r = 0; r < 8; ++r) { acc0[r] = bs[0]; acc1[r] = bs[1]; }

#pragma unroll
        for (int kc = 0; kc < KCH; ++kc) {
            HFrag a;
            const _Float16* rowp = &hbuf[frow][kc * 32 + kb8];
            a.h8[0] = *(const v8h*)(rowp);
            a.h8[1] = *(const v8h*)(rowp + 16);
            acc0 = __builtin_amdgcn_wmma_f32_16x16x32_f16(
                false, a.v, false, bfrag[0][kc].v, (short)0, acc0, false, false);
            acc1 = __builtin_amdgcn_wmma_f32_16x16x32_f16(
                false, a.v, false, bfrag[1][kc].v, (short)0, acc1, false, false);
        }

#pragma unroll
        for (int r = 0; r < 8; ++r) {
            gbuf[mtop + r][wv * 32 + frow]      = acc0[r];
            gbuf[mtop + r][wv * 32 + 16 + frow] = acc1[r];
        }
        __syncthreads();

        // pointwise recurrence: i,f,g,o -> c,h   (hardware trans ops)
#pragma unroll
        for (int q = 0; q < 4; ++q) {
            const int e  = tid + 512 * q;       // 0..2047
            const int br = e >> 7;              // batch row in slice
            const int j  = e & 127;             // hidden unit
            const float gi = gbuf[br][j];
            const float gf = gbuf[br][128 + j];
            const float gg = gbuf[br][256 + j];
            const float go = gbuf[br][384 + j];
            const float cn = sigf(gf) * cst[q] + sigf(gi) * tanhf_fast(gg);
            cst[q] = cn;
            const float hn = sigf(go) * tanhf_fast(cn);
            const _Float16 hh = (_Float16)hn;
            hbuf[br][j] = hh;                   // feeds next step's A fragments
            yout[((size_t)(bbase + br) * TDIM + t) * HDIM + j] = hh;
        }
        __syncthreads();
    }
}

// ---------------------------------------------------------------------------
// Kernel 4: length mask + FC (128 -> 7).  One thread per (b,t).
//   Reference masks lstm_out first, then adds fc_b (bias survives masking).
// ---------------------------------------------------------------------------
__global__ __launch_bounds__(256)
void fc_mask_kernel(const _Float16* __restrict__ y1,     // (B,T,128) f16
                    const int* __restrict__ lengths,     // (B)
                    const float* __restrict__ fc_w,      // (7,128)
                    const float* __restrict__ fc_b,      // (7)
                    float* __restrict__ out)             // (B,T,7)
{
    const int idx = blockIdx.x * blockDim.x + threadIdx.x;   // b*T + t
    const int b = idx / TDIM;
    const int t = idx - b * TDIM;

    float acc[CDIM];
#pragma unroll
    for (int c = 0; c < CDIM; ++c) acc[c] = 0.0f;

    if (t < lengths[b]) {
        const _Float16* hrow = y1 + (size_t)idx * HDIM;
#pragma unroll 4
        for (int j = 0; j < HDIM; ++j) {
            const float hv = (float)hrow[j];
#pragma unroll
            for (int c = 0; c < CDIM; ++c) acc[c] += fc_w[c * HDIM + j] * hv;
        }
    }
    float* orow = out + (size_t)idx * CDIM;
#pragma unroll
    for (int c = 0; c < CDIM; ++c) orow[c] = acc[c] + fc_b[c];
}

// ---------------------------------------------------------------------------
extern "C" void kernel_launch(void* const* d_in, const int* in_sizes, int n_in,
                              void* d_out, int out_size, void* d_ws, size_t ws_size,
                              hipStream_t stream) {
    const float* x       = (const float*)d_in[0];
    const int*   lengths = (const int*)  d_in[1];
    const float* conv_w  = (const float*)d_in[2];
    const float* conv_b  = (const float*)d_in[3];
    const float* mlp_w1  = (const float*)d_in[4];
    const float* mlp_b1  = (const float*)d_in[5];
    const float* mlp_w2  = (const float*)d_in[6];
    const float* mlp_b2  = (const float*)d_in[7];
    const float* ln_g    = (const float*)d_in[8];
    const float* ln_b    = (const float*)d_in[9];
    const float* w_ih0   = (const float*)d_in[10];
    const float* w_hh0   = (const float*)d_in[11];
    const float* b0      = (const float*)d_in[12];
    const float* w_ih1   = (const float*)d_in[13];
    const float* w_hh1   = (const float*)d_in[14];
    const float* b1      = (const float*)d_in[15];
    const float* fc_w    = (const float*)d_in[16];
    const float* fc_b    = (const float*)d_in[17];

    // workspace: tilde_X f16 (8 MB) | y0 f16 (64 MB) | y1 f16 (64 MB)
    char* ws = (char*)d_ws;
    _Float16* tx = (_Float16*)ws;
    _Float16* y0 = (_Float16*)(ws + (size_t)BDIM * TDIM * 16 * sizeof(_Float16));
    _Float16* y1 = (_Float16*)(ws + (size_t)BDIM * TDIM * (16 + HDIM) * sizeof(_Float16));

    const int nbt = BDIM * TDIM;

    frontend_kernel<<<nbt / 256, 256, 0, stream>>>(
        x, conv_w, conv_b, mlp_w1, mlp_b1, mlp_w2, mlp_b2, ln_g, ln_b, tx);

    // layer 0: K = 128(h) + 16(x) -> pad to 160 = 5 chunks of 32
    lstm_layer_kernel<5, 16><<<4, 512, 0, stream>>>(tx, w_ih0, w_hh0, b0, y0);

    // layer 1: K = 128(h) + 128(x) = 256 = 8 chunks of 32
    lstm_layer_kernel<8, 128><<<4, 512, 0, stream>>>(y0, w_ih1, w_hh1, b1, y1);

    fc_mask_kernel<<<nbt / 256, 256, 0, stream>>>(y1, lengths, fc_w, fc_b, (float*)d_out);
}
